// EnhancedGCN_68822555951628
// MI455X (gfx1250) — matrix operand order repaired and checked by reading, not compiled, not run on hardware
//
#include <hip/hip_runtime.h>
#include <hip/hip_bf16.h>

#define NN   100000
#define EE   1600000
#define FIN  1433
#define HID  64
#define NHEADS 2
#define NCLS 7
#define OUTC 14   // NHEADS*NCLS

typedef __attribute__((ext_vector_type(16))) _Float16 v16h;
typedef __attribute__((ext_vector_type(8)))  float    v8f;
typedef __attribute__((ext_vector_type(4)))  unsigned u32x4;
typedef __attribute__((ext_vector_type(8)))  int      i32x8;
typedef __attribute__((ext_vector_type(4)))  int      i32x4;

__device__ __forceinline__ unsigned f2ord(float f) {
  unsigned i = __float_as_uint(f);
  return (i & 0x80000000u) ? ~i : (i | 0x80000000u);
}
__device__ __forceinline__ float ord2f(unsigned k) {
  return (k & 0x80000000u) ? __uint_as_float(k & 0x7fffffffu) : __uint_as_float(~k);
}
__device__ __forceinline__ float lrelu02(float v) { return v > 0.0f ? v : 0.2f * v; }

// ---------------------------------------------------------------------------
// Tensor Data Mover: async DMA of a [16 rows x 32 cols] fp32 tile of a
// row-major [nrows, strideK] matrix into LDS. 2-D descriptor per ISA 8.3/8.4:
//   group0: count=1 | lds_addr | global_addr | type=2
//   group1: data_size=4B, tensor_dim0=remK (K-tail reads OOB -> zero-padded),
//           tensor_dim1=remRows, tile=32x16, tensor_dim0_stride=strideK
// Tracked by TENSORcnt.
// ---------------------------------------------------------------------------
__device__ __forceinline__ void tdm_load_tile(const float* gptr, unsigned lds_off,
                                              int remK, int remRows, int strideK) {
  unsigned long long ga = (unsigned long long)gptr;
  u32x4 g0;
  g0.x = 1u;                                   // count=1, is_restore=0, gather=0
  g0.y = lds_off;                              // LDS byte address
  g0.z = (unsigned)ga;                         // global_addr[31:0]
  g0.w = (unsigned)((ga >> 32) & 0x1FFFFFFu) | 0x80000000u;  // addr[56:32] | type=2
  i32x8 g1;
  g1[0] = 0x20000;                                                  // data_size=2 (4B)
  g1[1] = (int)((unsigned)remK << 16);                              // tensor_dim0[15:0]
  g1[2] = (int)((((unsigned)remK >> 16) & 0xFFFFu) |
                ((unsigned)remRows << 16));                         // dim0[31:16]|dim1[15:0]
  g1[3] = (int)((((unsigned)remRows >> 16) & 0xFFFFu) | (32u << 16)); // dim1[31:16]|tile_dim0=32
  g1[4] = 16;                                                       // tile_dim1=16, tile_dim2=0
  g1[5] = strideK;                                                  // tensor_dim0_stride[31:0]
  g1[6] = 0;
  g1[7] = 0;
  i32x4 z4 = {0, 0, 0, 0};
#if __clang_major__ >= 23
  i32x8 z8 = {0, 0, 0, 0, 0, 0, 0, 0};
  __builtin_amdgcn_tensor_load_to_lds(g0, g1, z4, z4, z8, 0);
#else
  __builtin_amdgcn_tensor_load_to_lds(g0, g1, z4, z4, 0);
#endif
}

// ---------------------------------------------------------------------------
// Pack a row-major fp32 weight [K, Ncols] into per-lane WMMA B-fragments (f16).
//   frag[((kc*ntiles + nt)*32 + lane)*16 + i] = W[kc*32 + (lane>>4)*16 + i][nt*16 + (lane&15)]
// ---------------------------------------------------------------------------
__global__ void pack_wfrag(const float* __restrict__ W, int K, int Ncols,
                           int nchunks, int ntiles, _Float16* __restrict__ frag) {
  int idx = blockIdx.x * blockDim.x + threadIdx.x;
  int total = nchunks * ntiles * 32 * 16;
  if (idx >= total) return;
  int i    = idx & 15;
  int lane = (idx >> 4) & 31;
  int cn   = idx >> 9;               // kc*ntiles + nt
  int nt   = cn % ntiles;
  int kc   = cn / ntiles;
  int k    = kc * 32 + (lane >> 4) * 16 + i;
  int n    = nt * 16 + (lane & 15);
  float v  = (k < K && n < Ncols) ? W[(long)k * Ncols + n] : 0.0f;
  frag[idx] = (_Float16)v;
}

// ---------------------------------------------------------------------------
// WMMA GEMM: Y[nrows, ldy] = X[nrows, K] @ Wfrag (fp32 in, f16 WMMA, fp32 out).
// nrows must be a multiple of 16 (NN = 6250*16). 128 threads = 4 waves; block
// computes a 16-row stripe; wave w owns output columns [w*16, w*16+16).
// X tiles stream in via double-buffered TDM (tensor_load_to_lds), overlapped
// with WMMA compute; K-tail is zero-padded by the TDM OOB rule.
// ---------------------------------------------------------------------------
__global__ void gemm_wmma(const float* __restrict__ X, int nrows, int K, int nchunks,
                          const _Float16* __restrict__ Wfrag, int ntiles,
                          float* __restrict__ Y, int ldy) {
  __shared__ __align__(16) float sF[2][16 * 32];
  const int row0 = blockIdx.x * 16;
  const int t    = threadIdx.x;      // 0..127
  const int wave = t >> 5;
  const int lane = t & 31;
  const int half = lane >> 4;
  const int m    = lane & 15;
  const bool active = (wave < ntiles);
  const bool leader = (wave == 0);
  const int remRows = nrows - row0;
  const float* xrow = X + (long)row0 * K;

  if (leader)
    tdm_load_tile(xrow, (unsigned)(unsigned long long)&sF[0][0], K, remRows, K);

  v8f acc = {};
  for (int kc = 0; kc < nchunks; ++kc) {
    const int cur = kc & 1;
    if (leader) __builtin_amdgcn_s_wait_tensorcnt(0);   // buffer[cur] landed
    __syncthreads();
    if (leader && (kc + 1) < nchunks)                   // prefetch next chunk
      tdm_load_tile(xrow + (kc + 1) * 32,
                    (unsigned)(unsigned long long)&sF[(kc + 1) & 1][0],
                    K - (kc + 1) * 32, remRows, K);
    if (active) {
      // A fragment: row m, K = half*16+0..15 (matches packed B fragment order)
      const float* rp = &sF[cur][m * 32];
      float4 f0 = *(const float4*)(rp + half * 16);
      float4 f1 = *(const float4*)(rp + half * 16 + 4);
      float4 f2 = *(const float4*)(rp + half * 16 + 8);
      float4 f3 = *(const float4*)(rp + half * 16 + 12);
      union { v16h v; _Float16 e[16]; } a;
      a.e[0]  = (_Float16)f0.x; a.e[1]  = (_Float16)f0.y;
      a.e[2]  = (_Float16)f0.z; a.e[3]  = (_Float16)f0.w;
      a.e[4]  = (_Float16)f1.x; a.e[5]  = (_Float16)f1.y;
      a.e[6]  = (_Float16)f1.z; a.e[7]  = (_Float16)f1.w;
      a.e[8]  = (_Float16)f2.x; a.e[9]  = (_Float16)f2.y;
      a.e[10] = (_Float16)f2.z; a.e[11] = (_Float16)f2.w;
      a.e[12] = (_Float16)f3.x; a.e[13] = (_Float16)f3.y;
      a.e[14] = (_Float16)f3.z; a.e[15] = (_Float16)f3.w;
      v16h b = *(const v16h*)&Wfrag[((kc * ntiles + wave) * 32 + lane) * 16];
      acc = __builtin_amdgcn_wmma_f32_16x16x32_f16(false, a.v, false, b,
                                                   (short)0, acc, false, false);
    }
    __syncthreads();
  }
  if (active) {
    // D layout: acc[r] -> (M = r + half*8, N = wave*16 + (lane&15)); no guards,
    // 32-bit indexing (nrows multiple of 16, indices < 2^31).
    int base = (row0 + half * 8) * ldy + wave * 16 + m;
    #pragma unroll
    for (int r = 0; r < 8; ++r) Y[base + r * ldy] = acc[r];
  }
}

// ---------------------------------------------------------------------------
// Degree / normalization
// ---------------------------------------------------------------------------
__global__ void deg_init(float* deg) {
  int i = blockIdx.x * blockDim.x + threadIdx.x;
  if (i < NN) deg[i] = 1.0f;         // self-loop contribution
}
__global__ void deg_edges(const int* __restrict__ dst, float* deg) {
  int e = blockIdx.x * blockDim.x + threadIdx.x;
  if (e < EE) atomicAdd(&deg[dst[e]], 1.0f);
}
__global__ void dinv_kernel(const float* __restrict__ deg, float* dinv) {
  int i = blockIdx.x * blockDim.x + threadIdx.x;
  if (i < NN) dinv[i] = rsqrtf(deg[i]);
}

// ---------------------------------------------------------------------------
// GCN propagation: out[dst] += H[src] * dinv[src]*dinv[dst], incl. self loops.
// 16 threads per edge, float4 per thread (coalesced 256B gather per edge).
// ---------------------------------------------------------------------------
__global__ void gcn_agg(const int* __restrict__ src, const int* __restrict__ dst,
                        const float* __restrict__ dinv, const float* __restrict__ H,
                        float* __restrict__ out) {
  long tid = (long)blockIdx.x * blockDim.x + threadIdx.x;
  if (tid >= (long)(EE + NN) * 16) return;
  int e = (int)(tid >> 4);
  int j = ((int)tid & 15) * 4;
  int s, d;
  if (e < EE) { s = src[e]; d = dst[e]; } else { s = d = e - EE; }
  float nrm = dinv[s] * dinv[d];
  float4 h4 = *(const float4*)&H[(long)s * HID + j];
  float* o = &out[(long)d * HID + j];
  atomicAdd(o + 0, h4.x * nrm);
  atomicAdd(o + 1, h4.y * nrm);
  atomicAdd(o + 2, h4.z * nrm);
  atomicAdd(o + 3, h4.w * nrm);
}

__global__ void bias_relu(float* __restrict__ h, const float* __restrict__ b) {
  long i = (long)blockIdx.x * blockDim.x + threadIdx.x;
  if (i >= (long)NN * HID) return;
  float v = h[i] + b[(int)(i & (HID - 1))];
  h[i] = fmaxf(v, 0.0f);
}

// ---------------------------------------------------------------------------
// GAT head
// ---------------------------------------------------------------------------
__global__ void gat_prep(const float* __restrict__ G,
                         const float* __restrict__ att_src, const float* __restrict__ att_dst,
                         float* __restrict__ asrc, float* __restrict__ adst,
                         unsigned* __restrict__ mkey, float* __restrict__ ssum) {
  int n = blockIdx.x * blockDim.x + threadIdx.x;
  if (n >= NN) return;
  #pragma unroll
  for (int h = 0; h < NHEADS; ++h) {
    float s = 0.0f, d = 0.0f;
    #pragma unroll
    for (int c = 0; c < NCLS; ++c) {
      float g = G[(long)n * 16 + h * NCLS + c];
      s += g * att_src[h * NCLS + c];
      d += g * att_dst[h * NCLS + c];
    }
    asrc[n * 2 + h] = s;
    adst[n * 2 + h] = d;
    mkey[n * 2 + h] = f2ord(-3.0e38f);
    ssum[n * 2 + h] = 0.0f;
  }
}

__global__ void gat_max(const int* __restrict__ src, const int* __restrict__ dst,
                        const float* __restrict__ asrc, const float* __restrict__ adst,
                        unsigned* __restrict__ mkey) {
  long tid = (long)blockIdx.x * blockDim.x + threadIdx.x;
  if (tid >= (long)(EE + NN) * 2) return;
  int e = (int)(tid >> 1), h = (int)tid & 1;
  int s, d;
  if (e < EE) { s = src[e]; d = dst[e]; } else { s = d = e - EE; }
  float v = lrelu02(asrc[s * 2 + h] + adst[d * 2 + h]);
  atomicMax(&mkey[d * 2 + h], f2ord(v));
}

__global__ void gat_expsum(const int* __restrict__ src, const int* __restrict__ dst,
                           const float* __restrict__ asrc, const float* __restrict__ adst,
                           const unsigned* __restrict__ mkey, float* __restrict__ ssum) {
  long tid = (long)blockIdx.x * blockDim.x + threadIdx.x;
  if (tid >= (long)(EE + NN) * 2) return;
  int e = (int)(tid >> 1), h = (int)tid & 1;
  int s, d;
  if (e < EE) { s = src[e]; d = dst[e]; } else { s = d = e - EE; }
  float v = lrelu02(asrc[s * 2 + h] + adst[d * 2 + h]);
  float m = ord2f(mkey[d * 2 + h]);
  atomicAdd(&ssum[d * 2 + h], expf(v - m));
}

__global__ void gat_aggregate(const int* __restrict__ src, const int* __restrict__ dst,
                              const float* __restrict__ asrc, const float* __restrict__ adst,
                              const unsigned* __restrict__ mkey, const float* __restrict__ ssum,
                              const float* __restrict__ G, float* __restrict__ out) {
  long tid = (long)blockIdx.x * blockDim.x + threadIdx.x;
  if (tid >= (long)(EE + NN) * 2) return;
  int e = (int)(tid >> 1), h = (int)tid & 1;
  int s, d;
  if (e < EE) { s = src[e]; d = dst[e]; } else { s = d = e - EE; }
  float v = lrelu02(asrc[s * 2 + h] + adst[d * 2 + h]);
  float m = ord2f(mkey[d * 2 + h]);
  float alpha = expf(v - m) / (ssum[d * 2 + h] + 1e-16f);
  #pragma unroll
  for (int c = 0; c < NCLS; ++c) {
    atomicAdd(&out[(long)d * 16 + h * NCLS + c],
              G[(long)s * 16 + h * NCLS + c] * alpha);
  }
}

__global__ void final_logsoftmax(const float* __restrict__ agg, const float* __restrict__ bg,
                                 float* __restrict__ out) {
  int n = blockIdx.x * blockDim.x + threadIdx.x;
  if (n >= NN) return;
  float v[OUTC];
  float mx = -3.0e38f;
  #pragma unroll
  for (int i = 0; i < OUTC; ++i) {
    v[i] = agg[(long)n * 16 + i] + bg[i];
    mx = fmaxf(mx, v[i]);
  }
  float sum = 0.0f;
  #pragma unroll
  for (int i = 0; i < OUTC; ++i) sum += expf(v[i] - mx);
  float ls = logf(sum);
  #pragma unroll
  for (int i = 0; i < OUTC; ++i) out[(long)n * OUTC + i] = v[i] - mx - ls;
}

// ---------------------------------------------------------------------------
static inline int cdiv(long a, int b) { return (int)((a + b - 1) / b); }

extern "C" void kernel_launch(void* const* d_in, const int* in_sizes, int n_in,
                              void* d_out, int out_size, void* d_ws, size_t ws_size,
                              hipStream_t stream) {
  const float* x       = (const float*)d_in[0];
  const int*   ei      = (const int*)d_in[1];
  const float* W1      = (const float*)d_in[2];
  const float* b1      = (const float*)d_in[3];
  const float* W2      = (const float*)d_in[4];
  const float* b2      = (const float*)d_in[5];
  const float* Wg      = (const float*)d_in[6];
  const float* att_src = (const float*)d_in[7];
  const float* att_dst = (const float*)d_in[8];
  const float* bg      = (const float*)d_in[9];
  float* out = (float*)d_out;

  const int* src = ei;        // edge_index[0]
  const int* dst = ei + EE;   // edge_index[1]

  // ---- workspace carving (floats unless noted) ----
  float* ws    = (float*)d_ws;
  float* bufA  = ws;                         // [NN,64] — XW1 / T2 / G(ld16)
  float* bufB  = bufA + (long)NN * 64;       // [NN,64] — agg1/h1, agg2/h2
  float* gagg  = bufB + (long)NN * 64;       // [NN,16] — GAT aggregate
  float* dinv  = gagg + (long)NN * 16;       // [NN]
  float* deg   = dinv + NN;                  // [NN]
  float* asrc  = deg + NN;                   // [NN,2]
  float* adst  = asrc + 2L * NN;             // [NN,2]
  unsigned* mkey = (unsigned*)(adst + 2L * NN); // [NN,2]
  float* ssum  = (float*)mkey + 2L * NN;     // [NN,2]
  _Float16* w1f = (_Float16*)(ssum + 2L * NN);  // 45*4*512
  _Float16* w2f = w1f + 45 * 4 * 512;           // 2*4*512
  _Float16* wgf = w2f + 2 * 4 * 512;            // 2*1*512

  const int B = 256;
  const int nblk16 = cdiv((long)(EE + NN) * 16, B);
  const int nblk2  = cdiv((long)(EE + NN) * 2, B);

  // normalization
  deg_init<<<cdiv(NN, B), B, 0, stream>>>(deg);
  deg_edges<<<cdiv(EE, B), B, 0, stream>>>(dst, deg);
  dinv_kernel<<<cdiv(NN, B), B, 0, stream>>>(deg, dinv);

  // weight fragment packing (tiny; resident in L2 afterwards)
  pack_wfrag<<<cdiv(45 * 4 * 512, B), B, 0, stream>>>(W1, FIN, HID, 45, 4, w1f);
  pack_wfrag<<<cdiv(2 * 4 * 512, B), B, 0, stream>>>(W2, HID, HID, 2, 4, w2f);
  pack_wfrag<<<cdiv(2 * 1 * 512, B), B, 0, stream>>>(Wg, HID, OUTC, 2, 1, wgf);

  // ---- GCN layer 1 ----
  hipMemsetAsync(bufB, 0, (size_t)NN * 64 * sizeof(float), stream);
  gemm_wmma<<<NN / 16, 128, 0, stream>>>(x, NN, FIN, 45, w1f, 4, bufA, 64);
  gcn_agg<<<nblk16, B, 0, stream>>>(src, dst, dinv, bufA, bufB);
  bias_relu<<<cdiv((long)NN * 64, B), B, 0, stream>>>(bufB, b1);

  // ---- GCN layer 2 ----
  gemm_wmma<<<NN / 16, 128, 0, stream>>>(bufB, NN, HID, 2, w2f, 4, bufA, 64);
  hipMemsetAsync(bufB, 0, (size_t)NN * 64 * sizeof(float), stream);
  gcn_agg<<<nblk16, B, 0, stream>>>(src, dst, dinv, bufA, bufB);
  bias_relu<<<cdiv((long)NN * 64, B), B, 0, stream>>>(bufB, b2);

  // ---- GAT head ----
  gemm_wmma<<<NN / 16, 128, 0, stream>>>(bufB, NN, HID, 2, wgf, 1, bufA, 16);
  gat_prep<<<cdiv(NN, B), B, 0, stream>>>(bufA, att_src, att_dst, asrc, adst, mkey, ssum);
  gat_max<<<nblk2, B, 0, stream>>>(src, dst, asrc, adst, mkey);
  hipMemsetAsync(gagg, 0, (size_t)NN * 16 * sizeof(float), stream);
  gat_expsum<<<nblk2, B, 0, stream>>>(src, dst, asrc, adst, mkey, ssum);
  gat_aggregate<<<nblk2, B, 0, stream>>>(src, dst, asrc, adst, mkey, ssum, bufA, gagg);
  final_logsoftmax<<<cdiv(NN, B), B, 0, stream>>>(gagg, bg, out);
}
